// MixedContextLoss_82952998355860
// MI455X (gfx1250) — matrix operand order repaired and checked by reading, not compiled, not run on hardware
//
#include <hip/hip_runtime.h>
#include <hip/hip_bf16.h>
#include <cstdint>
#include <cstddef>

// ---------------------------------------------------------------------------
// MixedContextLoss for MI455X (gfx1250, wave32, WMMA)
//   Hard-negative mining Gram matrix in split-bf16 (hi+lo, 3 WMMAs per tile)
//   accumulated f32 via v_wmma_f32_16x16x32_bf16. LDS ping-pong double buffer
//   + register prefetch: 1 barrier / k-step, global latency hidden by WMMAs.
//   Row-argmin fused into the epilogue (packed u64 global atomic-min);
//   final distances recomputed exactly in fp32.
// ---------------------------------------------------------------------------

typedef __attribute__((ext_vector_type(16))) __bf16 v16bf;
typedef __attribute__((ext_vector_type(8)))  float  v8f;

#define NB        8192      // batch
#define ND        2048      // feature dim
#define NTHREADS  256       // 8 waves of 32
#define TILE      128       // output tile (rows == cols)
#define KSTEP     32        // K per WMMA bf16 step
#define LDSTRIDE  40        // bf16 elems per LDS row (32 + 8 pad, bank-friendly)
#define TILESZ    (TILE * LDSTRIDE)   // elems in one LDS tile
#define BUFSZ     (4 * TILESZ)        // Ah|Al|Ph|Pl for one buffer

#define C_THETA   1.15f
#define C_DELTA   5.0f
#define C_GAMMA   0.5f
#define C_EPS     1e-6f

// order-preserving float -> uint32 (monotone: a<b  <=>  ford(a)<ford(b))
__device__ __forceinline__ unsigned int ford(float f) {
  unsigned int b = __float_as_uint(f);
  return (b & 0x80000000u) ? ~b : (b | 0x80000000u);
}

// split one fp32 into bf16 hi (truncated prefix) + bf16 lo (RNE of residual)
__device__ __forceinline__ void split2(float f, unsigned short& h, unsigned short& l) {
  unsigned int b = __float_as_uint(f);
  h = (unsigned short)(b >> 16);                       // truncation: hi is exact prefix
  float hf = __uint_as_float((unsigned int)h << 16);
  float r  = f - hf;                                   // exact residual
  unsigned int rb = __float_as_uint(r);
  unsigned int lsb = (rb >> 16) & 1u;
  l = (unsigned short)((rb + 0x7FFFu + lsb) >> 16);    // RNE bf16
}

// store 4 consecutive (hi,lo) bf16 pairs from a float4 into the two LDS tiles
__device__ __forceinline__ void store_split4(unsigned short* __restrict__ hiA,
                                             unsigned short* __restrict__ loA,
                                             int off, float4 v) {
  unsigned short h0, h1, h2, h3, l0, l1, l2, l3;
  split2(v.x, h0, l0); split2(v.y, h1, l1); split2(v.z, h2, l2); split2(v.w, h3, l3);
  *(unsigned int*)(hiA + off)     = (unsigned int)h0 | ((unsigned int)h1 << 16);
  *(unsigned int*)(hiA + off + 2) = (unsigned int)h2 | ((unsigned int)h3 << 16);
  *(unsigned int*)(loA + off)     = (unsigned int)l0 | ((unsigned int)l1 << 16);
  *(unsigned int*)(loA + off + 2) = (unsigned int)l2 | ((unsigned int)l3 << 16);
}

// A-fragment (16x32 bf16): lane<16 -> row m, K {0..7,16..23}; lane>=16 -> K {8..15,24..31}.
__device__ __forceinline__ v16bf load_fragA(const unsigned short* p) {
  struct alignas(16) U { uint4 a, b; } u;
  u.a = *(const uint4*)p;
  u.b = *(const uint4*)(p + 16);
  return __builtin_bit_cast(v16bf, u);
}

// B-fragment (32x16 bf16, B = P^T): lane<16 -> col n, K {0..7,16..23}; lane>=16 -> K {8..15,24..31}.
__device__ __forceinline__ v16bf load_fragB(const unsigned short* p) {
  struct alignas(16) U { uint4 a, b; } u;
  u.a = *(const uint4*)p;
  u.b = *(const uint4*)(p + 16);
  return __builtin_bit_cast(v16bf, u);
}

// ---------------------------------------------------------------------------
// Kernel 1: pnorm[j] = ||y_p[j]||^2 ; init packed argmin slots ; zero output
// ---------------------------------------------------------------------------
__global__ __launch_bounds__(NTHREADS) void init_kernel(
    const float* __restrict__ yp, float* __restrict__ pnorm,
    unsigned long long* __restrict__ packed, float* __restrict__ out) {
  const int row = blockIdx.x;
  const float* p = yp + (size_t)row * ND;
  float s = 0.f;
  for (int c = threadIdx.x; c < ND; c += NTHREADS) { float v = p[c]; s += v * v; }
  __shared__ float red[NTHREADS];
  red[threadIdx.x] = s;
  __syncthreads();
  for (int off = NTHREADS / 2; off > 0; off >>= 1) {
    if (threadIdx.x < off) red[threadIdx.x] += red[threadIdx.x + off];
    __syncthreads();
  }
  if (threadIdx.x == 0) {
    pnorm[row]  = red[0];
    packed[row] = 0xFFFFFFFFFFFFFFFFull;
    if (row == 0) out[0] = 0.f;
  }
}

// ---------------------------------------------------------------------------
// Kernel 2: split-bf16 WMMA Gram tile + fused masked row-argmin
//   score(i,j) = ||p_j||^2 - 2 * (y_a_i + eps) . y_p_j      (row-const dropped)
// ---------------------------------------------------------------------------
__global__ __launch_bounds__(NTHREADS, 1) void mine_kernel(
    const float* __restrict__ ya, const float* __restrict__ yp,
    const int* __restrict__ targets, const float* __restrict__ pnorm,
    unsigned long long* __restrict__ packed) {
  // two ping-pong buffers, each holding Ah | Al | Ph | Pl
  __shared__ __align__(16) unsigned short lds[2 * BUFSZ];

  const int tid  = threadIdx.x;
  const int lane = tid & 31;
  const int wave = tid >> 5;
  const int wi   = wave & 3;   // 4 row-groups of 32 rows
  const int wj   = wave >> 2;  // 2 col-groups of 64 cols
  const int m    = lane & 15;
  const int half = lane >> 4;

  const int ibase = blockIdx.y * TILE;
  const int jbase = blockIdx.x * TILE;

  // staging geometry (loop-invariant): 8 float4 per 32-col row slice
  const int rowBase = tid >> 3;        // 0..31, +32 per r
  const int c4      = (tid & 7) * 4;   // 0,4,...,28
  const float* aPtr = ya + (size_t)(ibase + rowBase) * ND + c4;
  const float* pPtr = yp + (size_t)(jbase + rowBase) * ND + c4;
  const int ldsOff  = rowBase * LDSTRIDE + c4;   // +32*LDSTRIDE per r

  v8f acc[2][4];
#pragma unroll
  for (int ci = 0; ci < 2; ++ci)
#pragma unroll
    for (int cj = 0; cj < 4; ++cj) acc[ci][cj] = (v8f)(0.0f);

  float4 aReg[4], pReg[4];

  // ---- prologue: load k-chunk 0 and stage into buffer 0 ----
#pragma unroll
  for (int r = 0; r < 4; ++r) {
    aReg[r] = *(const float4*)(aPtr + (size_t)(32 * r) * ND);
    pReg[r] = *(const float4*)(pPtr + (size_t)(32 * r) * ND);
  }
#pragma unroll
  for (int r = 0; r < 4; ++r) {
    float4 av = aReg[r];
    av.x += C_EPS; av.y += C_EPS; av.z += C_EPS; av.w += C_EPS;   // a = y_a + eps
    store_split4(lds + 0 * TILESZ, lds + 1 * TILESZ, ldsOff + 32 * r * LDSTRIDE, av);
    store_split4(lds + 2 * TILESZ, lds + 3 * TILESZ, ldsOff + 32 * r * LDSTRIDE, pReg[r]);
  }
  __syncthreads();

  // fragment base offsets within a buffer (elements)
  const int aFragOff = (wi * 32 + m) * LDSTRIDE + half * 8;   // +16*LDSTRIDE per ci
  const int bFragOff = (wj * 64 + m) * LDSTRIDE + half * 8;   // +16*LDSTRIDE per cj

  int bufOff = 0;
  for (int k0 = 0; k0 < ND; k0 += KSTEP) {
    // ---- prefetch next k-chunk into registers (latency hidden by WMMAs) ----
    const bool more = (k0 + KSTEP) < ND;
    if (more) {
#pragma unroll
      for (int r = 0; r < 4; ++r) {
        aReg[r] = *(const float4*)(aPtr + (size_t)(32 * r) * ND + (k0 + KSTEP));
        pReg[r] = *(const float4*)(pPtr + (size_t)(32 * r) * ND + (k0 + KSTEP));
      }
    }

    // ---- fragments from current buffer ----
    const unsigned short* Ah = lds + bufOff + 0 * TILESZ;
    const unsigned short* Al = lds + bufOff + 1 * TILESZ;
    const unsigned short* Ph = lds + bufOff + 2 * TILESZ;
    const unsigned short* Pl = lds + bufOff + 3 * TILESZ;

    v16bf ahi[2], alo[2], bhi[4], blo[4];
#pragma unroll
    for (int ci = 0; ci < 2; ++ci) {
      ahi[ci] = load_fragA(Ah + aFragOff + ci * 16 * LDSTRIDE);
      alo[ci] = load_fragA(Al + aFragOff + ci * 16 * LDSTRIDE);
    }
#pragma unroll
    for (int cj = 0; cj < 4; ++cj) {
      bhi[cj] = load_fragB(Ph + bFragOff + cj * 16 * LDSTRIDE);
      blo[cj] = load_fragB(Pl + bFragOff + cj * 16 * LDSTRIDE);
    }

    // ---- 3-term split product: hi*hi + hi*lo + lo*hi (lo*lo ~ 2^-16, dropped) ----
#pragma unroll
    for (int ci = 0; ci < 2; ++ci)
#pragma unroll
      for (int cj = 0; cj < 4; ++cj) {
        acc[ci][cj] = __builtin_amdgcn_wmma_f32_16x16x32_bf16(
            false, ahi[ci], false, bhi[cj], (short)0, acc[ci][cj], false, false);
        acc[ci][cj] = __builtin_amdgcn_wmma_f32_16x16x32_bf16(
            false, ahi[ci], false, blo[cj], (short)0, acc[ci][cj], false, false);
        acc[ci][cj] = __builtin_amdgcn_wmma_f32_16x16x32_bf16(
            false, alo[ci], false, bhi[cj], (short)0, acc[ci][cj], false, false);
      }

    // ---- stage prefetched chunk into the other buffer ----
    if (more) {
      unsigned short* nAh = lds + (bufOff ^ BUFSZ) + 0 * TILESZ;
      unsigned short* nAl = lds + (bufOff ^ BUFSZ) + 1 * TILESZ;
      unsigned short* nPh = lds + (bufOff ^ BUFSZ) + 2 * TILESZ;
      unsigned short* nPl = lds + (bufOff ^ BUFSZ) + 3 * TILESZ;
#pragma unroll
      for (int r = 0; r < 4; ++r) {
        float4 av = aReg[r];
        av.x += C_EPS; av.y += C_EPS; av.z += C_EPS; av.w += C_EPS;
        store_split4(nAh, nAl, ldsOff + 32 * r * LDSTRIDE, av);
        store_split4(nPh, nPl, ldsOff + 32 * r * LDSTRIDE, pReg[r]);
      }
    }
    bufOff ^= BUFSZ;
    __syncthreads();   // one barrier per k-step
  }

  // ---- fused masked argmin epilogue ----
  // C/D layout: vgpr v of lane L holds element (row = v + 8*(L>>4), col = L&15).
  unsigned int cols[4];
  float        pn[4];
  int          tcol[4];
#pragma unroll
  for (int cj = 0; cj < 4; ++cj) {
    const int col = jbase + wj * 64 + cj * 16 + m;
    cols[cj] = (unsigned int)col;
    pn[cj]   = pnorm[col];
    tcol[cj] = targets[col];
  }

#pragma unroll
  for (int ci = 0; ci < 2; ++ci) {
#pragma unroll
    for (int v = 0; v < 8; ++v) {
      const int rowg = ibase + wi * 32 + ci * 16 + v + 8 * half;
      const int trow = targets[rowg];
      unsigned long long best = 0xFFFFFFFFFFFFFFFFull;
#pragma unroll
      for (int cj = 0; cj < 4; ++cj) {
        const float s = pn[cj] - 2.0f * acc[ci][cj][v];
        const unsigned int u = (trow == tcol[cj]) ? 0xFFFFFFFFu : ford(s);
        const unsigned long long cand =
            ((unsigned long long)u << 32) | (unsigned long long)cols[cj];
        best = cand < best ? cand : best;
      }
      // min across the 16 lanes that hold this row's 16 columns
#pragma unroll
      for (int off = 8; off > 0; off >>= 1) {
        unsigned long long o = __shfl_xor(best, off, 16);
        best = o < best ? o : best;
      }
      if (m == 0) atomicMin(&packed[rowg], best);
    }
  }
}

// ---------------------------------------------------------------------------
// Kernel 3: gather negative, exact fp32 distances, stable loss, mean-reduce
// ---------------------------------------------------------------------------
__device__ __forceinline__ float log_sigmoid(float x) {
  return fminf(x, 0.f) - log1pf(expf(-fabsf(x)));
}

__global__ __launch_bounds__(NTHREADS) void loss_kernel(
    const float* __restrict__ ya, const float* __restrict__ yp,
    const unsigned long long* __restrict__ packed, float* __restrict__ out) {
  const int i = blockIdx.x;
  const int j = (int)(packed[i] & 0xFFFFFFFFull);
  const float* a = ya + (size_t)i * ND;
  const float* p = yp + (size_t)i * ND;
  const float* n = yp + (size_t)j * ND;
  float dp = 0.f, dn = 0.f;
  for (int c = threadIdx.x; c < ND; c += NTHREADS) {
    const float av = a[c];
    const float x  = av - p[c] + C_EPS;
    const float y  = av - n[c] + C_EPS;
    dp += x * x;
    dn += y * y;
  }
  __shared__ float rp[NTHREADS];
  __shared__ float rn[NTHREADS];
  rp[threadIdx.x] = dp;
  rn[threadIdx.x] = dn;
  __syncthreads();
  for (int off = NTHREADS / 2; off > 0; off >>= 1) {
    if (threadIdx.x < off) {
      rp[threadIdx.x] += rp[threadIdx.x + off];
      rn[threadIdx.x] += rn[threadIdx.x + off];
    }
    __syncthreads();
  }
  if (threadIdx.x == 0) {
    const float d_p   = sqrtf(rp[0]);
    const float d_n   = sqrtf(rn[0]);
    const float theta = C_GAMMA * (d_p + d_n) * 0.5f + (1.0f - C_GAMMA) * C_THETA;
    const float inv2d = 1.0f / (2.0f * C_DELTA);
    const float loss  = -(inv2d * log_sigmoid(2.0f * C_DELTA * (theta - d_p)) +
                          inv2d * log_sigmoid(2.0f * C_DELTA * (d_n - theta)));
    atomicAdd(out, loss * (1.0f / (float)NB));
  }
}

// ---------------------------------------------------------------------------
extern "C" void kernel_launch(void* const* d_in, const int* in_sizes, int n_in,
                              void* d_out, int out_size, void* d_ws, size_t ws_size,
                              hipStream_t stream) {
  (void)in_sizes; (void)n_in; (void)out_size; (void)ws_size;
  const float* ya = (const float*)d_in[0];
  const float* yp = (const float*)d_in[1];
  const int*   tg = (const int*)d_in[2];
  float*       out = (float*)d_out;

  float* pnorm = (float*)d_ws;                                         // 32 KB
  unsigned long long* packed =
      (unsigned long long*)((char*)d_ws + (size_t)NB * sizeof(float)); // 64 KB

  init_kernel<<<NB, NTHREADS, 0, stream>>>(yp, pnorm, packed, out);

  dim3 grid(NB / TILE, NB / TILE);
  mine_kernel<<<grid, NTHREADS, 0, stream>>>(ya, yp, tg, pnorm, packed);

  loss_kernel<<<NB, NTHREADS, 0, stream>>>(ya, yp, packed, out);
}
